// GIN_51702816309753
// MI455X (gfx1250) — matrix operand order repaired and checked by reading, not compiled, not run on hardware
//
#include <hip/hip_runtime.h>
#include <hip/hip_bf16.h>
#include <math.h>

// ---------------- problem constants ----------------
#define NN      100000      // nodes
#define NE      3200000     // edges
#define NG      1000        // graphs
#define C       128         // in/hid channels
#define OUTC    10
#define BN_EPS  1e-5f

typedef __attribute__((ext_vector_type(16))) _Float16 v16h;
typedef __attribute__((ext_vector_type(8)))  _Float16 v8h;
typedef __attribute__((ext_vector_type(8)))  float    v8f;

// ---------------- small utility kernels ----------------
__global__ void zerof_k(float* __restrict__ p, int n) {
    int t = blockIdx.x * blockDim.x + threadIdx.x;
    if (t < n) p[t] = 0.0f;
}

// agg = x  (float4 vectorized; n4 = NN*C/4)
__global__ void copy4_k(const float* __restrict__ src, float* __restrict__ dst) {
    int t = blockIdx.x * blockDim.x + threadIdx.x;   // exactly NN*C/4 threads
    reinterpret_cast<float4*>(dst)[t] = reinterpret_cast<const float4*>(src)[t];
}

// one wave (32 lanes) per edge; each lane moves 4 channels with f32 atomics
__global__ void edge_scatter_k(const float* __restrict__ x,
                               const int* __restrict__ src,
                               const int* __restrict__ dst,
                               float* __restrict__ agg) {
    int t = blockIdx.x * blockDim.x + threadIdx.x;
    int e  = t >> 5;
    int ln = t & 31;
    if (e >= NE) return;
    int s = src[e];
    int d = dst[e];
    float4 v = reinterpret_cast<const float4*>(x + (size_t)s * C)[ln];
    float* p = agg + (size_t)d * C + ln * 4;
    atomicAdd(p + 0, v.x);
    atomicAdd(p + 1, v.y);
    atomicAdd(p + 2, v.z);
    atomicAdd(p + 3, v.w);
}

// transpose+convert the four 128x128 weight matrices to f16 [n][k]
__global__ void wconv_k(const float* __restrict__ W1a, const float* __restrict__ W1b,
                        const float* __restrict__ W2a, const float* __restrict__ W2b,
                        _Float16* __restrict__ WT) {
    int t = blockIdx.x * blockDim.x + threadIdx.x;   // 4*128*128 threads
    int m = t >> 14;          // which matrix
    int i = t & 16383;        // i = k*128 + n
    int k = i >> 7;
    int n = i & 127;
    const float* W = (m == 0) ? W1a : (m == 1) ? W1b : (m == 2) ? W2a : W2b;
    WT[m * 16384 + n * 128 + k] = (_Float16)W[i];
}

// ---------------- fused MLP (2 GEMMs) with WMMA ----------------
// Computes one 16x16 output tile per wave:  D = A(16x128) * WT^T(128x16)
// A lives in LDS as f16 row-major [16][128]; WT is f16 [n][k] so each lane's
// B fragment is 16 contiguous halves.
__device__ inline v8f wmma_row_gemm(const _Float16* aT,        // LDS 16x128 f16
                                    const _Float16* __restrict__ WT, // global [128][128] f16, [n][k]
                                    int ln, int n0) {
    v8f acc = {};
    const int m   = ln & 15;
    const int n   = n0 + (ln & 15);
    const int hiA = (ln & 16) ? 16 : 0;
    const int hiB = (ln & 16) ? 16 : 0;
#pragma unroll
    for (int ks = 0; ks < 4; ++ks) {
        const int kA = ks * 32 + hiA;
        v8h alo = *(const v8h*)(aT + m * 128 + kA);
        v8h ahi = *(const v8h*)(aT + m * 128 + kA + 8);
        v16h a = __builtin_shufflevector(alo, ahi,
                                         0,1,2,3,4,5,6,7,8,9,10,11,12,13,14,15);
        const int kB = ks * 32 + hiB;
        v16h b = *(const v16h*)(WT + n * 128 + kB);
        acc = __builtin_amdgcn_wmma_f32_16x16x32_f16(
                  false, a, false, b, (short)0, acc, false, false);
    }
    return acc;
}

__global__ __launch_bounds__(256)
void gin_mlp_k(const float* __restrict__ agg,
               const _Float16* __restrict__ WaT, const float* __restrict__ ba,
               const _Float16* __restrict__ WbT, const float* __restrict__ bb,
               float* __restrict__ hOut,
               float* __restrict__ gSum, float* __restrict__ gSumSq) {
    __shared__ __align__(16) _Float16 aTile[16 * 128];
    __shared__ __align__(16) _Float16 h1Tile[16 * 128];
    __shared__ float lsum[128];
    __shared__ float lsq[128];

    const int tid = threadIdx.x;
    const int ln  = tid & 31;
    const int w   = tid >> 5;
    const int r0  = blockIdx.x * 16;      // 6250 blocks * 16 rows = 100000

    // stage A tile (f32 -> f16) into LDS; zero BN partials
    for (int idx = tid; idx < 16 * 128; idx += 256)
        aTile[idx] = (_Float16)agg[r0 * 128 + idx];
    if (tid < 128) { lsum[tid] = 0.0f; lsq[tid] = 0.0f; }
    __syncthreads();

    const int n0    = w * 16;
    const int n     = n0 + (ln & 15);
    const int mBase = (ln & 16) ? 8 : 0;

    // GEMM 1: relu(A @ Wa + ba) -> h1Tile (f16, LDS)
    v8f c1 = wmma_row_gemm(aTile, WaT, ln, n0);
    {
        float bias = ba[n];
#pragma unroll
        for (int r = 0; r < 8; ++r) {
            float v = c1[r] + bias;
            v = v > 0.0f ? v : 0.0f;
            h1Tile[(mBase + r) * 128 + n] = (_Float16)v;
        }
    }
    __syncthreads();

    // GEMM 2: h1 @ Wb + bb -> global hOut; accumulate BN stats
    v8f c2 = wmma_row_gemm(h1Tile, WbT, ln, n0);
    {
        float bias = bb[n];
        float s = 0.0f, sq = 0.0f;
#pragma unroll
        for (int r = 0; r < 8; ++r) {
            float v = c2[r] + bias;
            hOut[(r0 + mBase + r) * 128 + n] = v;
            s  += v;
            sq += v * v;
        }
        atomicAdd(&lsum[n], s);
        atomicAdd(&lsq[n], sq);
    }
    __syncthreads();
    if (tid < 128) {
        atomicAdd(&gSum[tid],   lsum[tid]);
        atomicAdd(&gSumSq[tid], lsq[tid]);
    }
}

// per-channel: scale = gamma * rsqrt(var+eps); shift = beta - mean*scale
__global__ void bn_finalize_k(const float* __restrict__ sums,
                              const float* __restrict__ gamma,
                              const float* __restrict__ beta,
                              float* __restrict__ scsh) {
    int c = threadIdx.x;           // 128 threads
    const float invN = 1.0f / (float)NN;
    float mean = sums[c] * invN;
    float var  = sums[128 + c] * invN - mean * mean;
    float sc   = gamma[c] * rsqrtf(var + BN_EPS);
    scsh[c]       = sc;
    scsh[128 + c] = beta[c] - mean * sc;
}

// y = relu(h * scale[c] + shift[c]), float4 vectorized
__global__ void bn_relu_k(const float* __restrict__ h,
                          const float* __restrict__ scsh,
                          float* __restrict__ y) {
    int t = blockIdx.x * blockDim.x + threadIdx.x;   // NN*C/4 threads
    int q = t & 31;                                  // channel-quad index
    float4 hv = reinterpret_cast<const float4*>(h)[t];
    float4 sc = reinterpret_cast<const float4*>(scsh)[q];
    float4 sh = reinterpret_cast<const float4*>(scsh + 128)[q];
    float4 o;
    o.x = fmaxf(hv.x * sc.x + sh.x, 0.0f);
    o.y = fmaxf(hv.y * sc.y + sh.y, 0.0f);
    o.z = fmaxf(hv.z * sc.z + sh.z, 0.0f);
    o.w = fmaxf(hv.w * sc.w + sh.w, 0.0f);
    reinterpret_cast<float4*>(y)[t] = o;
}

// pooled[batch[i]] += y[i]
__global__ void pool_k(const float* __restrict__ y,
                       const int* __restrict__ batch,
                       float* __restrict__ pooled) {
    int t = blockIdx.x * blockDim.x + threadIdx.x;   // NN*32 threads
    int node = t >> 5;
    int q    = t & 31;
    int g = batch[node];
    float4 v = reinterpret_cast<const float4*>(y + (size_t)node * C)[q];
    float* p = pooled + (size_t)g * C + q * 4;
    atomicAdd(p + 0, v.x);
    atomicAdd(p + 1, v.y);
    atomicAdd(p + 2, v.z);
    atomicAdd(p + 3, v.w);
}

// logits = pooled @ Wlin + blin ; log_softmax over 10 classes
__global__ void head_k(const float* __restrict__ pooled,
                       const float* __restrict__ Wlin,
                       const float* __restrict__ blin,
                       float* __restrict__ out) {
    int g = blockIdx.x * blockDim.x + threadIdx.x;
    if (g >= NG) return;
    float logit[OUTC];
#pragma unroll
    for (int o = 0; o < OUTC; ++o) logit[o] = blin[o];
    for (int k = 0; k < C; ++k) {
        float p = pooled[g * C + k];
#pragma unroll
        for (int o = 0; o < OUTC; ++o) logit[o] += p * Wlin[k * OUTC + o];
    }
    float mx = logit[0];
#pragma unroll
    for (int o = 1; o < OUTC; ++o) mx = fmaxf(mx, logit[o]);
    float se = 0.0f;
#pragma unroll
    for (int o = 0; o < OUTC; ++o) se += expf(logit[o] - mx);
    float lse = mx + logf(se);
#pragma unroll
    for (int o = 0; o < OUTC; ++o) out[g * OUTC + o] = logit[o] - lse;
}

// ---------------- launcher ----------------
extern "C" void kernel_launch(void* const* d_in, const int* in_sizes, int n_in,
                              void* d_out, int out_size, void* d_ws, size_t ws_size,
                              hipStream_t stream) {
    const float* x     = (const float*)d_in[0];
    const int*   ei    = (const int*)  d_in[1];   // [2, NE] flat: src then dst
    const int*   batch = (const int*)  d_in[2];
    const float* W1a = (const float*)d_in[3];
    const float* b1a = (const float*)d_in[4];
    const float* W1b = (const float*)d_in[5];
    const float* b1b = (const float*)d_in[6];
    const float* g1  = (const float*)d_in[7];
    const float* be1 = (const float*)d_in[8];
    const float* W2a = (const float*)d_in[9];
    const float* b2a = (const float*)d_in[10];
    const float* W2b = (const float*)d_in[11];
    const float* b2b = (const float*)d_in[12];
    const float* g2  = (const float*)d_in[13];
    const float* be2 = (const float*)d_in[14];
    const float* Wlin = (const float*)d_in[15];
    const float* blin = (const float*)d_in[16];
    float* out = (float*)d_out;

    char* ws = (char*)d_ws;
    const size_t FEAT = (size_t)NN * C * sizeof(float);   // 51,200,000 B
    float*    agg    = (float*)(ws);
    float*    hbuf   = (float*)(ws + FEAT);
    float*    ybuf   = (float*)(ws + 2 * FEAT);
    _Float16* WT     = (_Float16*)(ws + 3 * FEAT);        // 4 * 16384 f16
    float*    sums   = (float*)(ws + 3 * FEAT + 131072);  // 256 f32 (sum, sumsq)
    float*    scsh   = sums + 256;                        // 256 f32 (scale, shift)
    float*    pooled = scsh + 256;                        // NG*C f32

    const int n4 = NN * C / 4;                // 3,200,000
    const int copyBlocks = n4 / 256;          // 12,500
    const int edgeBlocks = (NE * 32) / 256;   // 400,000
    const int mlpBlocks  = NN / 16;           // 6,250

    // weight prep + pooled zero
    wconv_k<<<(4 * 16384) / 256, 256, 0, stream>>>(W1a, W1b, W2a, W2b, WT);
    zerof_k<<<(NG * C + 255) / 256, 256, 0, stream>>>(pooled, NG * C);

    // ---- layer 1 ----
    copy4_k<<<copyBlocks, 256, 0, stream>>>(x, agg);
    edge_scatter_k<<<edgeBlocks, 256, 0, stream>>>(x, ei, ei + NE, agg);
    zerof_k<<<1, 256, 0, stream>>>(sums, 256);
    gin_mlp_k<<<mlpBlocks, 256, 0, stream>>>(agg, WT, b1a, WT + 16384, b1b,
                                             hbuf, sums, sums + 128);
    bn_finalize_k<<<1, 128, 0, stream>>>(sums, g1, be1, scsh);
    bn_relu_k<<<copyBlocks, 256, 0, stream>>>(hbuf, scsh, ybuf);

    // ---- layer 2 ----
    copy4_k<<<copyBlocks, 256, 0, stream>>>(ybuf, agg);
    edge_scatter_k<<<edgeBlocks, 256, 0, stream>>>(ybuf, ei, ei + NE, agg);
    zerof_k<<<1, 256, 0, stream>>>(sums, 256);
    gin_mlp_k<<<mlpBlocks, 256, 0, stream>>>(agg, WT + 32768, b2a, WT + 49152, b2b,
                                             hbuf, sums, sums + 128);
    bn_finalize_k<<<1, 128, 0, stream>>>(sums, g2, be2, scsh);
    bn_relu_k<<<copyBlocks, 256, 0, stream>>>(hbuf, scsh, ybuf);

    // ---- readout ----
    pool_k<<<copyBlocks, 256, 0, stream>>>(ybuf, batch, pooled);
    head_k<<<(NG + 255) / 256, 256, 0, stream>>>(pooled, Wlin, blin, out);

    (void)in_sizes; (void)n_in; (void)out_size; (void)ws_size;
}
